// PINN1_61864708932294
// MI455X (gfx1250) — matrix operand order repaired and checked by reading, not compiled, not run on hardware
//
#include <hip/hip_runtime.h>
#include <hip/hip_bf16.h>

typedef __attribute__((ext_vector_type(16))) _Float16 v16h;
typedef __attribute__((ext_vector_type(8)))  _Float16 v8h;
typedef __attribute__((ext_vector_type(8)))  float    v8f;

#define NW 4                      // waves per block (wave32)
#define SMEM_BYTES 118720

// Branchless tanh: v_exp_f32 + v_rcp_f32, saturates to +/-1, no EXEC churn.
__device__ __forceinline__ float fast_tanhf(float z) {
  float e = __expf(2.f * z);                    // v_exp_f32 (scaled)
  float r = __builtin_amdgcn_rcpf(e + 1.f);     // v_rcp_f32
  return 1.f - 2.f * r;
}

// A-fragment (16x32 f16, ISA layout): lane half = lane>>4, row m = lane&15.
// elements 0..7  <- K = base + half*8 .. +7
// elements 8..15 <- K = base + half*8 + 16 .. +7
__device__ __forceinline__ v16h ld_afrag(const _Float16* p) {
  v16h r;
  ((v8h*)&r)[0] = *(const v8h*)(p);
  ((v8h*)&r)[1] = *(const v8h*)(p + 16);
  return r;
}
// B-fragment (32x16 f16): lane holds column n, K = base + half*8 .. +7 and base + half*8 + 16 .. +7.
__device__ __forceinline__ v16h ld_bfrag(const _Float16* p) {
  v16h r;
  ((v8h*)&r)[0] = *(const v8h*)(p);
  ((v8h*)&r)[1] = *(const v8h*)(p + 16);
  return r;
}

__global__ void __launch_bounds__(NW * 32)
pinn_irk_kernel(const float* __restrict__ W0, const float* __restrict__ b0,
                const float* __restrict__ W1, const float* __restrict__ b1,
                const float* __restrict__ W2, const float* __restrict__ b2,
                const float* __restrict__ W3, const float* __restrict__ b3,
                const float* __restrict__ W4, const float* __restrict__ b4,
                const float* __restrict__ W5, const float* __restrict__ b5,
                const float* __restrict__ x,
                const float* __restrict__ lam1p, const float* __restrict__ lam2p,
                const float* __restrict__ alpha, const float* __restrict__ dtp,
                float* __restrict__ out)
{
  extern __shared__ __align__(16) char smem[];
  _Float16* sW   = (_Float16*)smem;          // [4][64*64]  W1..W4 transposed: sW[l][n*64+k] = W[k][n]
  _Float16* sW5t = sW + 4 * 4096;            // [112*64]    sW5t[n*64+k] = W5[k][n]
  _Float16* sAl  = sW5t + 112 * 64;          // [112*128]   sAl[n*128+k] = alpha[n][k]  (B of f@alpha^T)
  _Float16* sAct = sAl + 112 * 128;          // [NW][3][16*64] activations (u,ux,uxx)
  _Float16* sF   = sAct + NW * 3 * 1024;     // [NW][16*128]   f tile for IRK matmul
  float*    sBh  = (float*)(sF + NW * 2048); // [4][64] hidden biases
  float*    sB5  = sBh + 4 * 64;             // [112]
  float*    sW0  = sB5 + 112;                // [64]
  float*    sB0  = sW0 + 64;                 // [64]

  const int tid  = threadIdx.x;
  const int nthr = NW * 32;

  // ---- cooperative weight staging + f32->f16 conversion (tiny, once per block) ----
  const float* Whid[4] = {W1, W2, W3, W4};
  const float* Bhid[4] = {b1, b2, b3, b4};
  for (int i = tid; i < 64; i += nthr) {
    sW0[i] = (i < 50) ? W0[i] : 0.f;
    sB0[i] = (i < 50) ? b0[i] : 0.f;
  }
  for (int l = 0; l < 4; ++l) {
    const float* W = Whid[l];
    for (int idx = tid; idx < 64 * 64; idx += nthr) {
      int n = idx >> 6, k = idx & 63;
      sW[l * 4096 + idx] = (n < 50 && k < 50) ? (_Float16)W[k * 50 + n] : (_Float16)0.f;
    }
    for (int n = tid; n < 64; n += nthr)
      sBh[l * 64 + n] = (n < 50) ? Bhid[l][n] : 0.f;
  }
  for (int idx = tid; idx < 112 * 64; idx += nthr) {
    int n = idx >> 6, k = idx & 63;
    sW5t[idx] = (n < 100 && k < 50) ? (_Float16)W5[k * 100 + n] : (_Float16)0.f;
  }
  for (int n = tid; n < 112; n += nthr)
    sB5[n] = (n < 100) ? b5[n] : 0.f;
  for (int idx = tid; idx < 112 * 128; idx += nthr) {
    int n = idx >> 7, k = idx & 127;
    sAl[idx] = (n < 100 && k < 100) ? (_Float16)alpha[n * 100 + k] : (_Float16)0.f;
  }
  __syncthreads();

  const float lam1 = lam1p[0];
  const float el2  = __expf(lam2p[0]);
  const float dt   = dtp[0];

  const int wave = tid >> 5;
  const int lane = tid & 31;
  const int half = lane >> 4;    // A/B K-half, C row-half
  const int mrow = lane & 15;    // A row
  const int col  = lane & 15;    // B/C column within 16-chunk

  const long tile = (long)blockIdx.x * NW + wave;   // 16 samples per wave
  _Float16* act = sAct + wave * (3 * 1024);
  _Float16* fb  = sF + wave * 2048;

  // ---- layer 0: z = x*W0+b0 ; tangents z'=W0, z''=0 ----
  {
    float xm = x[tile * 16 + mrow];
    int cb = half * 32;
    #pragma unroll
    for (int c = 0; c < 32; ++c) {
      int n = cb + c;
      float w = sW0[n];
      float z = xm * w + sB0[n];
      float y = fast_tanhf(z);
      float g = 1.f - y * y;
      float yp = g * w;
      float ypp = -2.f * y * yp * w;
      act[0 * 1024 + mrow * 64 + n] = (_Float16)y;
      act[1 * 1024 + mrow * 64 + n] = (_Float16)yp;
      act[2 * 1024 + mrow * 64 + n] = (_Float16)ypp;
    }
  }
  asm volatile("s_wait_dscnt 0" ::: "memory");

  // ---- hidden layers W1..W4: 3 joint matmuls [16x64]@[64x64] + tanh chain rule ----
  for (int l = 0; l < 4; ++l) {
    v16h A[3][2];
    #pragma unroll
    for (int s = 0; s < 3; ++s)
      #pragma unroll
      for (int kc = 0; kc < 2; ++kc)
        A[s][kc] = ld_afrag(act + s * 1024 + mrow * 64 + kc * 32 + half * 8);

    const _Float16* Wl = sW + l * 4096;
    #pragma unroll
    for (int nc = 0; nc < 4; ++nc) {
      int gcol = nc * 16 + col;
      float bias = sBh[l * 64 + gcol];
      v8f au  = {bias, bias, bias, bias, bias, bias, bias, bias};
      v8f ap  = {};
      v8f app = {};
      #pragma unroll
      for (int kc = 0; kc < 2; ++kc) {
        v16h B = ld_bfrag(Wl + gcol * 64 + kc * 32 + half * 8);
        au  = __builtin_amdgcn_wmma_f32_16x16x32_f16(false, A[0][kc], false, B, (short)0, au,  false, false);
        ap  = __builtin_amdgcn_wmma_f32_16x16x32_f16(false, A[1][kc], false, B, (short)0, ap,  false, false);
        app = __builtin_amdgcn_wmma_f32_16x16x32_f16(false, A[2][kc], false, B, (short)0, app, false, false);
      }
      #pragma unroll
      for (int r = 0; r < 8; ++r) {
        int row = r + half * 8;
        float y = fast_tanhf(au[r]);
        float g = 1.f - y * y;
        float yp = g * ap[r];
        float ypp = g * app[r] - 2.f * y * yp * ap[r];
        act[0 * 1024 + row * 64 + gcol] = (_Float16)y;
        act[1 * 1024 + row * 64 + gcol] = (_Float16)yp;
        act[2 * 1024 + row * 64 + gcol] = (_Float16)ypp;
      }
    }
    asm volatile("s_wait_dscnt 0" ::: "memory");
  }

  // ---- output layer W5 (linear) -> u,u_x,u_xx ; f = -l1*u*ux + e^l2*uxx ----
  v16h A[3][2];
  #pragma unroll
  for (int s = 0; s < 3; ++s)
    #pragma unroll
    for (int kc = 0; kc < 2; ++kc)
      A[s][kc] = ld_afrag(act + s * 1024 + mrow * 64 + kc * 32 + half * 8);

  float ureg[7][8];
  #pragma unroll
  for (int nc = 0; nc < 7; ++nc) {
    int gcol = nc * 16 + col;
    float bias = sB5[gcol];
    v8f au  = {bias, bias, bias, bias, bias, bias, bias, bias};
    v8f ap  = {};
    v8f app = {};
    #pragma unroll
    for (int kc = 0; kc < 2; ++kc) {
      v16h B = ld_bfrag(sW5t + gcol * 64 + kc * 32 + half * 8);
      au  = __builtin_amdgcn_wmma_f32_16x16x32_f16(false, A[0][kc], false, B, (short)0, au,  false, false);
      ap  = __builtin_amdgcn_wmma_f32_16x16x32_f16(false, A[1][kc], false, B, (short)0, ap,  false, false);
      app = __builtin_amdgcn_wmma_f32_16x16x32_f16(false, A[2][kc], false, B, (short)0, app, false, false);
    }
    #pragma unroll
    for (int r = 0; r < 8; ++r) {
      int row = r + half * 8;
      float u = au[r];
      float f = -lam1 * u * ap[r] + el2 * app[r];
      ureg[nc][r] = u;
      fb[row * 128 + gcol] = (_Float16)f;
    }
  }
  { // zero-pad f columns 112..127 for the K=128 IRK matmul
    int gcol = 112 + col;
    #pragma unroll
    for (int r = 0; r < 8; ++r)
      fb[(r + half * 8) * 128 + gcol] = (_Float16)0.f;
  }
  asm volatile("s_wait_dscnt 0" ::: "memory");

  // ---- IRK stage: out = u - dt * (f @ alpha^T),  [16x128]@[128x112] ----
  v16h Fa[4];
  #pragma unroll
  for (int kc = 0; kc < 4; ++kc)
    Fa[kc] = ld_afrag(fb + mrow * 128 + kc * 32 + half * 8);

  #pragma unroll
  for (int nc = 0; nc < 7; ++nc) {
    int gcol = nc * 16 + col;
    v8f acc = {};
    #pragma unroll
    for (int kc = 0; kc < 4; ++kc) {
      v16h B = ld_bfrag(sAl + gcol * 128 + kc * 32 + half * 8);
      acc = __builtin_amdgcn_wmma_f32_16x16x32_f16(false, Fa[kc], false, B, (short)0, acc, false, false);
    }
    if (gcol < 100) {
      #pragma unroll
      for (int r = 0; r < 8; ++r) {
        long row = tile * 16 + (r + half * 8);
        out[row * 100 + gcol] = ureg[nc][r] - dt * acc[r];
      }
    }
  }
}

extern "C" void kernel_launch(void* const* d_in, const int* in_sizes, int n_in,
                              void* d_out, int out_size, void* d_ws, size_t ws_size,
                              hipStream_t stream) {
  const float* W0 = (const float*)d_in[0];
  const float* b0 = (const float*)d_in[1];
  const float* W1 = (const float*)d_in[2];
  const float* b1 = (const float*)d_in[3];
  const float* W2 = (const float*)d_in[4];
  const float* b2 = (const float*)d_in[5];
  const float* W3 = (const float*)d_in[6];
  const float* b3 = (const float*)d_in[7];
  const float* W4 = (const float*)d_in[8];
  const float* b4 = (const float*)d_in[9];
  const float* W5 = (const float*)d_in[10];
  const float* b5 = (const float*)d_in[11];
  const float* x  = (const float*)d_in[12];
  const float* l1 = (const float*)d_in[13];
  const float* l2 = (const float*)d_in[14];
  const float* al = (const float*)d_in[15];
  const float* dt = (const float*)d_in[16];
  float* out = (float*)d_out;

  const int N = in_sizes[12];           // 262144 collocation points
  const int grid = N / (16 * NW);       // one 16-sample tile per wave

  pinn_irk_kernel<<<grid, NW * 32, SMEM_BYTES, stream>>>(
      W0, b0, W1, b1, W2, b2, W3, b3, W4, b4, W5, b5,
      x, l1, l2, al, dt, out);
}
